// SGFMNet_69363721830734
// MI455X (gfx1250) — matrix operand order repaired and checked by reading, not compiled, not run on hardware
//
#include <hip/hip_runtime.h>
#include <math.h>

// ---------------- problem constants (from reference) ----------------
#define HID 256
#define LAYERS 4
#define ATOMD 100
#define TDIMC 128
#define NFQ 10
#define BG 128           // graphs
#define NNODE 32         // nodes per graph
#define GSZ 8            // group size
#define NTOT (BG*NNODE)  // 4096
#define DIN_E 581
#define KPAD1 608        // 581 padded to multiple of 32
#define TWO_PI_F 6.283185307179586f

// ---------------- WMMA types / helpers ----------------
typedef __attribute__((ext_vector_type(16))) __bf16 bf16x16;
typedef __attribute__((ext_vector_type(8)))  __bf16 bf16x8;
typedef __attribute__((ext_vector_type(8)))  float  f32x8;

__device__ __forceinline__ __bf16 f2bf(float f) {
  unsigned u = __builtin_bit_cast(unsigned, f);
  unsigned r = u + 0x7FFFu + ((u >> 16) & 1u);   // round-to-nearest-even
  unsigned short h = (unsigned short)(r >> 16);
  return __builtin_bit_cast(__bf16, h);
}
__device__ __forceinline__ float siluf(float x) { return x / (1.0f + __expf(-x)); }

// A-matrix 16x32 bf16 fragment. lane<16 holds row M=lane, K {c..c+15};
// lane>=16 holds row M=lane-16, K {c+16..c+31}.  row_chunk points at
// the lane's row at the K-chunk base; half = lane>>4.
__device__ __forceinline__ bf16x16 load_a_frag(const __bf16* row_chunk, int half) {
  const __bf16* q = row_chunk + half * 16;
  bf16x8 lo = *(const bf16x8*)(q);
  bf16x8 hi = *(const bf16x8*)(q + 8);
  bf16x16 r;
#pragma unroll
  for (int i = 0; i < 8; ++i) { r[i] = lo[i]; r[i + 8] = hi[i]; }
  return r;
}
// B-matrix 32x16 bf16 fragment: lane n (&15) = column N, halves split K 0..15/16..31.
// Since B = W^T, a B column is a contiguous W row: wrow = W + n*Kpad + kbase + half*16.
__device__ __forceinline__ bf16x16 load_b_frag(const __bf16* wrow) {
  bf16x8 lo = *(const bf16x8*)(wrow);
  bf16x8 hi = *(const bf16x8*)(wrow + 8);
  bf16x16 r;
#pragma unroll
  for (int i = 0; i < 8; ++i) { r[i] = lo[i]; r[i + 8] = hi[i]; }
  return r;
}
__device__ __forceinline__ f32x8 wmma_bf16(bf16x16 a, bf16x16 b, f32x8 c) {
  return __builtin_amdgcn_wmma_f32_16x16x32_bf16(false, a, false, b, (short)0, c,
                                                 false, false);
}
// Opaque barrier on a scalar offset: defeats LICM hoist-and-spill of the
// weight-fragment loads out of the block/layer loops (kept in an SGPR).
__device__ __forceinline__ int opaque_s(int v) {
  asm volatile("" : "+s"(v));
  return v;
}

// ---------------- workspace byte offsets ----------------
#define WS_LTL   0u                       // B*9 f32
#define WS_H0    8192u                    // N*256 f32 = 4 MB
#define WS_W1P   (WS_H0 + 4194304u)       // L*256*608 bf16
#define WS_W2P   (WS_W1P + 1245184u)      // L*256*256 bf16
#define WS_N1P   (WS_W2P + 524288u)       // L*256*512 bf16
#define WS_N2P   (WS_N1P + 1048576u)      // L*256*256 bf16

// ---------------- LDS byte offsets (main kernel) ----------------
#define SM_HF    0u        // 32*256 f32   (master h)
#define SM_HB    32768u    // 32*256 bf16
#define SM_EF1   49152u    // 64*256 bf16
#define SM_AGG   81920u    // 32*256 f32   (scatter-mean accumulator)
#define SM_XB    114688u   // 32*256 bf16  (agg_bf16 then nf1_bf16)
#define SM_TAIL  131072u   // 64*96 bf16
#define SM_FR    143360u   // 96 f32
#define SM_LTL   143744u   // 9 f32
#define SM_LAT   143780u   // 6 f32
#define SM_CRD   143804u   // 96 f32
#define SM_GF    144188u   // 256 f32
#define SMEM_BYTES 145216u

// ================= kernel 1: weight packing (f32 -> bf16, pad K) ===========
extern "C" __global__ void pack_weights_kernel(
    const float* __restrict__ ew1, const float* __restrict__ ew2,
    const float* __restrict__ nw1, const float* __restrict__ nw2,
    __bf16* __restrict__ w1p, __bf16* __restrict__ w2p,
    __bf16* __restrict__ n1p, __bf16* __restrict__ n2p) {
  int i = blockIdx.x * 256 + threadIdx.x;
  const int T1 = LAYERS * 256 * KPAD1;   // 622592
  const int T2 = LAYERS * 256 * 256;     // 262144
  const int T3 = LAYERS * 256 * 512;     // 524288
  if (i < T1) {
    int k = i % KPAD1, rl = i / KPAD1;   // rl = layer*256 + out_row
    w1p[i] = (k < DIN_E) ? f2bf(ew1[rl * DIN_E + k]) : f2bf(0.0f);
    return;
  }
  i -= T1;
  if (i < T2) { w2p[i] = f2bf(ew2[i]); return; }
  i -= T2;
  if (i < T3) { n1p[i] = f2bf(nw1[i]); return; }
  i -= T3;
  if (i < T2) { n2p[i] = f2bf(nw2[i]); return; }
}

// ================= kernel 2: 3x3 symmetric expm + L^T L ====================
__device__ __forceinline__ void mat3mul(float* __restrict__ P,
                                        const float* __restrict__ A,
                                        const float* __restrict__ B) {
#pragma unroll
  for (int i = 0; i < 3; ++i)
#pragma unroll
    for (int j = 0; j < 3; ++j)
      P[i * 3 + j] = A[i * 3 + 0] * B[0 * 3 + j] + A[i * 3 + 1] * B[1 * 3 + j] +
                     A[i * 3 + 2] * B[2 * 3 + j];
}

extern "C" __global__ void lattice_expm_kernel(
    const float* __restrict__ lattices, const float* __restrict__ k_mean,
    const float* __restrict__ k_std, const float* __restrict__ k_mask,
    const float* __restrict__ k_bias, float* __restrict__ ltl_out) {
  int b = blockIdx.x * blockDim.x + threadIdx.x;
  if (b >= BG) return;
  float y[6];
#pragma unroll
  for (int k = 0; k < 6; ++k) {
    float v = lattices[b * 6 + k] * k_std[k] + k_mean[k];
    y[k] = v * k_mask[k] + k_bias[k];
  }
  float S[9] = {y[0], y[3], y[4], y[3], y[1], y[5], y[4], y[5], y[2]};
  float nrm = 0.0f;
#pragma unroll
  for (int i = 0; i < 9; ++i) nrm += S[i] * S[i];
  nrm = sqrtf(nrm);
  int sh = 0;
  float sc = 1.0f;
  while (nrm * sc > 0.25f && sh < 30) { sc *= 0.5f; ++sh; }
  float X[9];
#pragma unroll
  for (int i = 0; i < 9; ++i) X[i] = S[i] * sc;
  float T[9] = {1, 0, 0, 0, 1, 0, 0, 0, 1};
  for (int k = 12; k >= 1; --k) {            // Horner Taylor of exp(X)
    float P[9];
    mat3mul(P, X, T);
    float ik = 1.0f / (float)k;
#pragma unroll
    for (int i = 0; i < 9; ++i)
      T[i] = ((i == 0) | (i == 4) | (i == 8) ? 1.0f : 0.0f) + P[i] * ik;
  }
  for (int q = 0; q < sh; ++q) {             // squaring
    float P[9];
    mat3mul(P, T, T);
#pragma unroll
    for (int i = 0; i < 9; ++i) T[i] = P[i];
  }
#pragma unroll
  for (int i = 0; i < 3; ++i)
#pragma unroll
    for (int j = 0; j < 3; ++j)
      ltl_out[b * 9 + i * 3 + j] = T[0 * 3 + i] * T[0 * 3 + j] +
                                   T[1 * 3 + i] * T[1 * 3 + j] +
                                   T[2 * 3 + i] * T[2 * 3 + j];
}

// ================= kernel 3: initial node embedding ========================
extern "C" __global__ __launch_bounds__(256) void embed_kernel(
    const float* __restrict__ t, const float* __restrict__ atom,
    const float* __restrict__ emb_w, const float* __restrict__ emb_b,
    const float* __restrict__ latent_w, const float* __restrict__ latent_b,
    float* __restrict__ h0) {
  __shared__ float hpre[HID];
  __shared__ float temb[TDIMC];
  int n = blockIdx.x, tid = threadIdx.x;
  {
    float s = emb_b[tid];
    const float* ar = atom + n * ATOMD;
    const float* wr = emb_w + tid * ATOMD;
    for (int k = 0; k < ATOMD; ++k) s += ar[k] * wr[k];
    hpre[tid] = s;
  }
  if (tid < TDIMC) {
    float tv = t[n >> 5];
    int j = (tid < 64) ? tid : tid - 64;
    float tf = __expf(-9.210340371976184f * (float)j * (1.0f / 64.0f));
    float ta = tv * tf;
    temb[tid] = (tid < 64) ? __cosf(ta) : __sinf(ta);
  }
  __syncthreads();
  float s = latent_b[tid];
  const float* wr = latent_w + tid * (HID + TDIMC);
  for (int k = 0; k < HID; ++k) s += hpre[k] * wr[k];
  for (int k = 0; k < TDIMC; ++k) s += temb[k] * wr[HID + k];
  h0[n * HID + tid] = s;
}

// ================= kernel 4: fused 4-layer message passing + heads =========
// one workgroup (256 threads = 8 wave32) per graph; all layers in LDS.
extern "C" __global__ __launch_bounds__(256) void crystal_mpnn_main(
    const float* __restrict__ h0, const float* __restrict__ ltl_g,
    const __bf16* __restrict__ w1p, const __bf16* __restrict__ w2p,
    const __bf16* __restrict__ n1p, const __bf16* __restrict__ n2p,
    const float* __restrict__ eb1, const float* __restrict__ eb2,
    const float* __restrict__ nb1, const float* __restrict__ nb2,
    const float* __restrict__ frac_g, const float* __restrict__ lat_g,
    const float* __restrict__ coord_w, const float* __restrict__ lattice_w,
    const float* __restrict__ Gmat, const int* __restrict__ invperm,
    float* __restrict__ out) {
  extern __shared__ char smem[];
  float*  hf   = (float*)(smem + SM_HF);
  __bf16* hb   = (__bf16*)(smem + SM_HB);
  __bf16* ef1  = (__bf16*)(smem + SM_EF1);
  float*  agg  = (float*)(smem + SM_AGG);
  __bf16* xb   = (__bf16*)(smem + SM_XB);
  __bf16* tail = (__bf16*)(smem + SM_TAIL);
  float*  fr   = (float*)(smem + SM_FR);
  float*  ltl  = (float*)(smem + SM_LTL);
  float*  lat  = (float*)(smem + SM_LAT);
  float*  crd  = (float*)(smem + SM_CRD);
  float*  gf   = (float*)(smem + SM_GF);

  const int b = blockIdx.x;
  const int tid = threadIdx.x;
  const int lane = tid & 31, wv = tid >> 5;
  const int col = lane & 15, half = (lane >> 4) & 1;

  for (int i = tid; i < NNODE * HID; i += 256) {
    float v = h0[b * NNODE * HID + i];
    hf[i] = v;
    hb[i] = f2bf(v);
  }
  for (int i = tid; i < NNODE * 3; i += 256) fr[i] = frac_g[b * NNODE * 3 + i];
  if (tid < 9) ltl[tid] = ltl_g[b * 9 + tid];
  if (tid < 6) lat[tid] = lat_g[b * 6 + tid];
  __syncthreads();

  for (int l = 0; l < LAYERS; ++l) {
    const float* B1 = eb1 + l * 256;
    const float* B2 = eb2 + l * 256;

    for (int i = tid; i < NNODE * HID; i += 256) agg[i] = 0.0f;
    __syncthreads();

    // -------- edge MLP over 16 blocks of 64 edges --------
    for (int blk = 0; blk < 16; ++blk) {
      // build per-edge tail [64][96]: lat(6) | sin/cos dist(60) | l_f(3) | pad
      if (tid < 64) {
        int e = blk * 64 + tid;
        int s = e >> 5, d = e & 31;
        __bf16* tr = tail + tid * 96;
        float fd[3];
#pragma unroll
        for (int k = 0; k < 3; ++k) {
          float z = TWO_PI_F * (fr[d * 3 + k] - fr[s * 3 + k]);
          fd[k] = atan2f(__sinf(z), __cosf(z)) * (1.0f / TWO_PI_F);
        }
#pragma unroll
        for (int k = 0; k < 6; ++k) tr[k] = f2bf(lat[k]);
#pragma unroll
        for (int k = 0; k < 3; ++k)
          for (int f = 0; f < NFQ; ++f) {
            float de = fd[k] * (TWO_PI_F * (float)f);
            tr[6 + k * NFQ + f] = f2bf(__sinf(de));
            tr[36 + k * NFQ + f] = f2bf(__cosf(de));
          }
        float lf[3], nn2 = 0.0f;
#pragma unroll
        for (int i = 0; i < 3; ++i) {
          lf[i] = ltl[i * 3 + 0] * fd[0] + ltl[i * 3 + 1] * fd[1] +
                  ltl[i * 3 + 2] * fd[2];
          nn2 += lf[i] * lf[i];
        }
        float inv = 1.0f / (sqrtf(nn2) + 1e-6f);
#pragma unroll
        for (int i = 0; i < 3; ++i) tr[66 + i] = f2bf(lf[i] * inv);
        for (int k = 69; k < 96; ++k) tr[k] = f2bf(0.0f);
      }
      __syncthreads();

      // GEMM1: ef1[64,256] = silu(ein[64,608] @ W1^T + B1); A gathered from LDS
      {
        // opaque per-iteration offset: keep weight loads inside this loop
        const __bf16* W1L = w1p + opaque_s(l * 256 * KPAD1);
        const int mt = wv >> 1, nh = wv & 1;
        const int e = blk * 64 + mt * 16 + col;
        const __bf16* asrc = hb + (e >> 5) * HID;
        const __bf16* adst = hb + (e & 31) * HID;
        const __bf16* atl = tail + (mt * 16 + col) * 96;
        f32x8 acc[8] = {};
        bf16x16 a = load_a_frag(asrc, half);
        for (int kb = 0; kb < 19; ++kb) {
          bf16x16 an = a;
          if (kb < 18) {
            int k1 = kb + 1;
            const __bf16* ap = (k1 < 8)    ? (asrc + k1 * 32)
                               : (k1 < 16) ? (adst + (k1 - 8) * 32)
                                           : (atl + (k1 - 16) * 32);
            an = load_a_frag(ap, half);
          }
          const __bf16* wb = W1L + kb * 32 + half * 16 + col * KPAD1 +
                             (nh * 8) * 16 * KPAD1;
          bf16x16 bc = load_b_frag(wb);
#pragma unroll
          for (int nt = 0; nt < 8; ++nt) {
            bf16x16 bn = bc;
            if (nt < 7) bn = load_b_frag(wb + (nt + 1) * 16 * KPAD1);
            acc[nt] = wmma_bf16(a, bc, acc[nt]);
            bc = bn;
          }
          a = an;
        }
#pragma unroll
        for (int nt = 0; nt < 8; ++nt) {
          int n = ((nh * 8 + nt) << 4) + col;
          float bias = B1[n];
#pragma unroll
          for (int r = 0; r < 8; ++r) {
            int row = mt * 16 + r + half * 8;
            ef1[row * HID + n] = f2bf(siluf(acc[nt][r] + bias));
          }
        }
      }
      __syncthreads();

      // GEMM2 + scatter-mean: each 16-edge tile lies in one src segment
      {
        const __bf16* W2L = w2p + opaque_s(l * 256 * 256);
        const int mt = wv >> 1, nh = wv & 1;
        const __bf16* arow = ef1 + (mt * 16 + col) * HID;
        f32x8 acc[8] = {};
        bf16x16 a = load_a_frag(arow, half);
        for (int kb = 0; kb < 8; ++kb) {
          bf16x16 an = a;
          if (kb < 7) an = load_a_frag(arow + (kb + 1) * 32, half);
          const __bf16* wb = W2L + kb * 32 + half * 16 + col * 256 +
                             (nh * 8) * 16 * 256;
          bf16x16 bc = load_b_frag(wb);
#pragma unroll
          for (int nt = 0; nt < 8; ++nt) {
            bf16x16 bn = bc;
            if (nt < 7) bn = load_b_frag(wb + (nt + 1) * 16 * 256);
            acc[nt] = wmma_bf16(a, bc, acc[nt]);
            bc = bn;
          }
          a = an;
        }
        const int srcn = (blk * 64 + mt * 16) >> 5;
#pragma unroll
        for (int nt = 0; nt < 8; ++nt) {
          int n = ((nh * 8 + nt) << 4) + col;
          float bias = B2[n];
          float ssum = 0.0f;
#pragma unroll
          for (int r = 0; r < 8; ++r) ssum += siluf(acc[nt][r] + bias);
          atomicAdd(&agg[srcn * HID + n], ssum);   // ds_add_f32
        }
      }
      __syncthreads();
    }

    // agg -> bf16 (scatter-mean: deg == 32 always)
    for (int i = tid; i < NNODE * HID; i += 256)
      xb[i] = f2bf(agg[i] * (1.0f / 32.0f));
    __syncthreads();

    // -------- node MLP: nf = silu(silu([h|agg] @ N1^T + nb1) @ N2^T + nb2)
    const float* NB1 = nb1 + l * 256;
    const float* NB2 = nb2 + l * 256;
    {
      const __bf16* N1L = n1p + opaque_s(l * 256 * 512);
      const int mt = wv & 1, ntg = (wv >> 1) * 4;
      const __bf16* ah = hb + (mt * 16 + col) * HID;
      const __bf16* aa = xb + (mt * 16 + col) * HID;
      f32x8 acc[4] = {};
      bf16x16 a = load_a_frag(ah, half);
      for (int kb = 0; kb < 16; ++kb) {
        bf16x16 an = a;
        if (kb < 15) {
          int k1 = kb + 1;
          const __bf16* ap = (k1 < 8) ? (ah + k1 * 32) : (aa + (k1 - 8) * 32);
          an = load_a_frag(ap, half);
        }
        const __bf16* wb = N1L + kb * 32 + half * 16 + col * 512 + ntg * 16 * 512;
        bf16x16 bc = load_b_frag(wb);
#pragma unroll
        for (int nt = 0; nt < 4; ++nt) {
          bf16x16 bn = bc;
          if (nt < 3) bn = load_b_frag(wb + (nt + 1) * 16 * 512);
          acc[nt] = wmma_bf16(a, bc, acc[nt]);
          bc = bn;
        }
        a = an;
      }
      __syncthreads();  // all reads of xb done; reuse it for nf1
#pragma unroll
      for (int nt = 0; nt < 4; ++nt) {
        int n = ((ntg + nt) << 4) + col;
        float bias = NB1[n];
#pragma unroll
        for (int r = 0; r < 8; ++r) {
          int row = mt * 16 + r + half * 8;
          xb[row * HID + n] = f2bf(siluf(acc[nt][r] + bias));
        }
      }
    }
    __syncthreads();
    {
      const __bf16* N2L = n2p + opaque_s(l * 256 * 256);
      const int mt = wv & 1, ntg = (wv >> 1) * 4;
      const __bf16* arow = xb + (mt * 16 + col) * HID;
      f32x8 acc[4] = {};
      bf16x16 a = load_a_frag(arow, half);
      for (int kb = 0; kb < 8; ++kb) {
        bf16x16 an = a;
        if (kb < 7) an = load_a_frag(arow + (kb + 1) * 32, half);
        const __bf16* wb = N2L + kb * 32 + half * 16 + col * 256 + ntg * 16 * 256;
        bf16x16 bc = load_b_frag(wb);
#pragma unroll
        for (int nt = 0; nt < 4; ++nt) {
          bf16x16 bn = bc;
          if (nt < 3) bn = load_b_frag(wb + (nt + 1) * 16 * 256);
          acc[nt] = wmma_bf16(a, bc, acc[nt]);
          bc = bn;
        }
        a = an;
      }
#pragma unroll
      for (int nt = 0; nt < 4; ++nt) {
        int n = ((ntg + nt) << 4) + col;
        float bias = NB2[n];
#pragma unroll
        for (int r = 0; r < 8; ++r) {
          int row = mt * 16 + r + half * 8;
          hf[row * HID + n] += siluf(acc[nt][r] + bias);   // residual
        }
      }
    }
    __syncthreads();
    for (int i = tid; i < NNODE * HID; i += 256) hb[i] = f2bf(hf[i]);
    __syncthreads();
  }

  // -------- heads --------
  {  // mean pool -> graph feature
    float s = 0.0f;
    for (int m = 0; m < NNODE; ++m) s += hf[m * HID + tid];
    gf[tid] = s * (1.0f / 32.0f);
  }
  __syncthreads();
  if (tid < 6) {  // lattice head
    float s = 0.0f;
    for (int c = 0; c < HID; ++c) s += gf[c] * lattice_w[tid * HID + c];
    out[NTOT * 3 + b * 6 + tid] = s;
  }
  if (tid < 96) {  // coord head
    int m = tid / 3, d2 = tid % 3;
    float s = 0.0f;
    for (int c = 0; c < HID; ++c) s += hf[m * HID + c] * coord_w[d2 * HID + c];
    crd[m * 3 + d2] = s;
  }
  __syncthreads();
  if (tid < 96) {  // equivariant split-permute-transform-average
    int nloc = tid / 3, j = tid % 3;
    float s = 0.0f;
    for (int g = 0; g < GSZ; ++g) {
      int gi = b * GSZ + g;
      int p = invperm[gi * NNODE + nloc];
      const float* Gr = Gmat + gi * 16 + j * 4;
      s += crd[p * 3 + 0] * Gr[0] + crd[p * 3 + 1] * Gr[1] +
           crd[p * 3 + 2] * Gr[2] + Gr[3];
    }
    out[(b * NNODE + nloc) * 3 + j] = s * (1.0f / (float)GSZ);
  }
}

// ================= host launcher ===========================================
extern "C" void kernel_launch(void* const* d_in, const int* in_sizes, int n_in,
                              void* d_out, int out_size, void* d_ws,
                              size_t ws_size, hipStream_t stream) {
  (void)in_sizes; (void)n_in; (void)out_size; (void)ws_size;
  const float* t        = (const float*)d_in[0];
  const float* atom     = (const float*)d_in[1];
  const float* frac     = (const float*)d_in[2];
  const float* lattices = (const float*)d_in[3];
  const float* G        = (const float*)d_in[4];
  const float* k_mean   = (const float*)d_in[5];
  const float* k_std    = (const float*)d_in[6];
  const float* k_mask   = (const float*)d_in[7];
  const float* k_bias   = (const float*)d_in[8];
  const float* emb_w    = (const float*)d_in[9];
  const float* emb_b    = (const float*)d_in[10];
  const float* latent_w = (const float*)d_in[11];
  const float* latent_b = (const float*)d_in[12];
  const float* edge_w1  = (const float*)d_in[13];
  const float* edge_b1  = (const float*)d_in[14];
  const float* edge_w2  = (const float*)d_in[15];
  const float* edge_b2  = (const float*)d_in[16];
  const float* node_w1  = (const float*)d_in[17];
  const float* node_b1  = (const float*)d_in[18];
  const float* node_w2  = (const float*)d_in[19];
  const float* node_b2  = (const float*)d_in[20];
  const float* coord_w  = (const float*)d_in[21];
  const float* lattice_w= (const float*)d_in[22];
  const int*   invperm  = (const int*)d_in[25];
  float* out = (float*)d_out;

  char* ws = (char*)d_ws;
  float*  ltl = (float*)(ws + WS_LTL);
  float*  h0  = (float*)(ws + WS_H0);
  __bf16* w1p = (__bf16*)(ws + WS_W1P);
  __bf16* w2p = (__bf16*)(ws + WS_W2P);
  __bf16* n1p = (__bf16*)(ws + WS_N1P);
  __bf16* n2p = (__bf16*)(ws + WS_N2P);

  static bool attr_set = false;
  if (!attr_set) {  // host-side launch config only; no effect on device work
    hipFuncSetAttribute((const void*)crystal_mpnn_main,
                        hipFuncAttributeMaxDynamicSharedMemorySize,
                        (int)SMEM_BYTES);
    attr_set = true;
  }

  const int packTotal = LAYERS * 256 * (KPAD1 + 256 + 512 + 256);  // 1671168
  hipLaunchKernelGGL(pack_weights_kernel, dim3((packTotal + 255) / 256),
                     dim3(256), 0, stream, edge_w1, edge_w2, node_w1, node_w2,
                     w1p, w2p, n1p, n2p);
  hipLaunchKernelGGL(lattice_expm_kernel, dim3(1), dim3(128), 0, stream,
                     lattices, k_mean, k_std, k_mask, k_bias, ltl);
  hipLaunchKernelGGL(embed_kernel, dim3(NTOT), dim3(256), 0, stream, t, atom,
                     emb_w, emb_b, latent_w, latent_b, h0);
  hipLaunchKernelGGL(crystal_mpnn_main, dim3(BG), dim3(256), SMEM_BYTES, stream,
                     h0, ltl, w1p, w2p, n1p, n2p, edge_b1, edge_b2, node_b1,
                     node_b2, frac, lattices, coord_w, lattice_w, G, invperm,
                     out);
}
